// LFADS_3272765080214
// MI455X (gfx1250) — matrix operand order repaired and checked by reading, not compiled, not run on hardware
//
#include <hip/hip_runtime.h>
#include <hip/hip_bf16.h>
#include <math.h>

// ---------------------------------------------------------------------------
// LFADS decoder scan, persistent-kernel formulation for gfx1250 (MI455X).
//   - 64 blocks x 256 threads (8 wave32), each block owns 16 batch rows.
//   - Recurrent states live in LDS (f32 master + f16 WMMA operand copy).
//   - All matmuls: v_wmma_f32_16x16x32_f16, weights pre-packed to the
//     B-fragment lane layout so the inner loop is pure b128 loads + WMMA.
//   - z/r gates accumulate X@W and H@W into the SAME WMMA accumulator
//     (D = A*B + C chaining) to cut VGPR pressure and avoid spills.
// ---------------------------------------------------------------------------

typedef __attribute__((ext_vector_type(16))) _Float16 v16h;
typedef __attribute__((ext_vector_type(8)))  _Float16 v8h;
typedef __attribute__((ext_vector_type(8)))  float    v8f;

#define BATCH   1024
#define T_STEPS 100
#define GEN     256
#define CON     128
#define CO      16
#define FAC     64
#define IC_DIM  64
#define CI2     128      // 2*CI_ENC_DIM
#define CONIN   192      // CI2 + FAC
#define OUTW    496      // GEN + CON + CO + CO + CO + FAC
#define CLIPV   5.0f

__device__ __forceinline__ v8f wmma16(v16h a, v16h b, v8f c) {
  return __builtin_amdgcn_wmma_f32_16x16x32_f16(false, a, false, b,
                                                (short)0, c, false, false);
}

// A-fragment (16x32 f16) from row-major f16 LDS buffer, per ISA layout:
// lanes 0-15 row M=lane, K = {k0..k0+7, k0+16..k0+23};
// lanes 16-31 row M=lane-16, K = {k0+8..k0+15, k0+24..k0+31}.
// Two aligned 16-byte LDS loads per fragment.
__device__ __forceinline__ v16h load_a(const _Float16* buf, int strideK,
                                       int k0, int lid) {
  const int r = lid & 15, hi = lid >> 4;
  v8h lo = *(const v8h*)(buf + r * strideK + k0 + hi * 8);
  v8h hh = *(const v8h*)(buf + r * strideK + k0 + 16 + hi * 8);
  v16h a;
#pragma unroll
  for (int i = 0; i < 8; ++i) { a[i] = lo[i]; a[i + 8] = hh[i]; }
  return a;
}

// B-fragment: pre-packed in global memory as [((nt*KT+kt)*32 + lane)*16 + j],
// one aligned 32-byte load per lane (2x global_load_b128, L2-resident).
__device__ __forceinline__ v16h load_b(const _Float16* pk, int KT, int nt,
                                       int kt, int lid) {
  return *(const v16h*)(pk + (((size_t)(nt * KT + kt) * 32 + lid) << 4));
}

__device__ __forceinline__ float sigm(float x) { return 1.f / (1.f + expf(-x)); }
__device__ __forceinline__ float clip5(float x) {
  return fminf(CLIPV, fmaxf(-CLIPV, x));
}

// ---------------------------------------------------------------------------
// Prologue 1: reciprocal row norms of W_fac (64 rows x 256 cols)
// ---------------------------------------------------------------------------
__global__ void fac_norm_kernel(const float* __restrict__ W,
                                float* __restrict__ rnorm) {
  __shared__ float red[256];
  const int row = blockIdx.x;
  float v = W[row * 256 + threadIdx.x];
  red[threadIdx.x] = v * v;
  __syncthreads();
  for (int s = 128; s > 0; s >>= 1) {
    if (threadIdx.x < s) red[threadIdx.x] += red[threadIdx.x + s];
    __syncthreads();
  }
  if (threadIdx.x == 0) rnorm[row] = 1.0f / sqrtf(red[0]);
}

// ---------------------------------------------------------------------------
// Prologue 2: pack W (outDim x inDim, f32, row-major, y = x @ W^T) into
// f16 B-fragments matching load_a's K order: column n = lane&15;
// half index j maps to k = kt*32 + (lane>>4)*8 + (j&7) + (j>>3)*16.
// K positions >= inDim are zero padding (used for the K=16 generator input).
// ---------------------------------------------------------------------------
__global__ void pack_wt_kernel(const float* __restrict__ W,
                               _Float16* __restrict__ dst, int outDim,
                               int inDim, int KT,
                               const float* __restrict__ rscale, int total) {
  const int idx = blockIdx.x * blockDim.x + threadIdx.x;
  if (idx >= total) return;
  const int j    = idx & 15;
  const int lane = (idx >> 4) & 31;
  const int tle  = idx >> 9;
  const int kt   = tle % KT;
  const int nt   = tle / KT;
  const int n = nt * 16 + (lane & 15);
  const int k = kt * 32 + (lane >> 4) * 8 + (j & 7) + (j >> 3) * 16;
  float v = (k < inDim) ? W[(size_t)n * inDim + k] : 0.f;
  if (rscale) v *= rscale[n];
  dst[idx] = (_Float16)v;
}

// ---------------------------------------------------------------------------
// Main persistent kernel: one block = 16 batch rows, loops the full scan.
// ---------------------------------------------------------------------------
__global__ __launch_bounds__(256) void lfads_persistent(
    const float* __restrict__ ic, const float* __restrict__ ci,
    const float* __restrict__ eps, const float* __restrict__ b_ic,
    const float* __restrict__ con_h0, const float* __restrict__ b_ih_gen,
    const float* __restrict__ b_ih_con, const float* __restrict__ b_co,
    const _Float16* __restrict__ pWic, const _Float16* __restrict__ pWihCon,
    const _Float16* __restrict__ pWhhCon, const _Float16* __restrict__ pWco,
    const _Float16* __restrict__ pWihGen, const _Float16* __restrict__ pWhhGen,
    const _Float16* __restrict__ pWfac, float* __restrict__ out) {
  __shared__ _Float16 sHg16[16 * GEN];    // generator state, f16 operand copy
  __shared__ float    sHg32[16 * GEN];    // generator state, f32 master
  __shared__ _Float16 sHc16[16 * CON];    // controller state f16
  __shared__ float    sHc32[16 * CON];    // controller state f32
  __shared__ _Float16 sX16[16 * CONIN];   // [ci_t | factor] f16 (also ic at init)
  __shared__ _Float16 sGi16[16 * 32];     // gen input, K padded 16->32
  __shared__ float    sCoM[16 * 16];
  __shared__ float    sCoS[16 * 16];
  __shared__ float    sGi32[16 * 16];
  __shared__ float    sFac[16 * FAC];

  const int tid = threadIdx.x;
  const int lid = tid & 31;
  const int wv  = tid >> 5;
  const int r0  = blockIdx.x * 16;
  const int col = lid & 15;
  const int mo  = (lid >= 16) ? 8 : 0;   // C/D frag: VGPR i -> row i (+8 hi half)

  // ---- init: zero gen-input pad, load ic_samp, broadcast con_h0 ----
  for (int i = tid; i < 16 * 32; i += 256) sGi16[i] = (_Float16)0.f;
  for (int i = tid; i < 16 * IC_DIM; i += 256) {
    int m = i >> 6, c = i & 63;
    sX16[m * CONIN + c] = (_Float16)ic[(size_t)(r0 + m) * IC_DIM + c];
  }
  for (int i = tid; i < 16 * CON; i += 256) {
    int c = i & 127;
    float v = con_h0[c];
    sHc32[i] = v;
    sHc16[i] = (_Float16)v;
  }
  __syncthreads();

  // ---- gen_init = ic @ W_ic^T + b_ic ----
  {
    float hs[2][8];
    for (int ct = 0; ct < 2; ++ct) {
      const int nt = 2 * wv + ct;
      v8f acc = {};
      for (int kt = 0; kt < 2; ++kt) {
        v16h a = load_a(sX16, CONIN, kt * 32, lid);
        acc = wmma16(a, load_b(pWic, 2, nt, kt, lid), acc);
      }
      const float bv = b_ic[nt * 16 + col];
#pragma unroll
      for (int i = 0; i < 8; ++i) hs[ct][i] = acc[i] + bv;
    }
    __syncthreads();
    for (int ct = 0; ct < 2; ++ct) {
      const int gcol = (2 * wv + ct) * 16 + col;
#pragma unroll
      for (int i = 0; i < 8; ++i) {
        const int m = i + mo;
        sHg32[m * GEN + gcol] = hs[ct][i];
        sHg16[m * GEN + gcol] = (_Float16)hs[ct][i];
      }
    }
    __syncthreads();
  }

  // ---- factor0 = gen_init @ W_fac_n^T ----
  if (wv < 4) {
    v8f acc = {};
    for (int kt = 0; kt < 8; ++kt) {
      v16h a = load_a(sHg16, GEN, kt * 32, lid);
      acc = wmma16(a, load_b(pWfac, 8, wv, kt, lid), acc);
    }
    const int gcol = wv * 16 + col;
#pragma unroll
    for (int i = 0; i < 8; ++i) {
      const int m = i + mo;
      sFac[m * FAC + gcol] = acc[i];
      sX16[m * CONIN + CI2 + gcol] = (_Float16)acc[i];
    }
  }
  __syncthreads();

  // ================= sequential scan over T =================
  for (int t = 0; t < T_STEPS; ++t) {
    // phase 0: stream ci_t into con-input buffer (cols 0..127)
    for (int i = tid; i < 16 * CI2; i += 256) {
      int m = i >> 7, c = i & 127;
      sX16[m * CONIN + c] =
          (_Float16)ci[((size_t)(r0 + m) * T_STEPS + t) * CI2 + c];
    }
    __syncthreads();

    // phase A: controller GRU. Wave w owns hidden cols [16w,16w+16).
    // z/r accumulators carry BOTH X@W and H@W contributions (WMMA chaining).
    {
      v8f az = {}, ar = {}, xn = {}, hn = {};
      for (int kt = 0; kt < 6; ++kt) {
        v16h a = load_a(sX16, CONIN, kt * 32, lid);
        az = wmma16(a, load_b(pWihCon, 6, wv, kt, lid), az);
        ar = wmma16(a, load_b(pWihCon, 6, 8 + wv, kt, lid), ar);
        xn = wmma16(a, load_b(pWihCon, 6, 16 + wv, kt, lid), xn);
      }
      for (int kt = 0; kt < 4; ++kt) {
        v16h a = load_a(sHc16, CON, kt * 32, lid);
        az = wmma16(a, load_b(pWhhCon, 4, wv, kt, lid), az);
        ar = wmma16(a, load_b(pWhhCon, 4, 8 + wv, kt, lid), ar);
        hn = wmma16(a, load_b(pWhhCon, 4, 16 + wv, kt, lid), hn);
      }
      const int gcol = wv * 16 + col;
      const float bz = b_ih_con[gcol];
      const float br = b_ih_con[CON + gcol];
      const float bn = b_ih_con[2 * CON + gcol];
      float hnew[8];
#pragma unroll
      for (int i = 0; i < 8; ++i) {
        const int m = i + mo;
        const float hold = sHc32[m * CON + gcol];
        const float z  = sigm(az[i] + bz);
        const float rr = sigm(ar[i] + br);
        const float nn = tanhf(xn[i] + bn + rr * hn[i]);
        hnew[i] = clip5(z * hold + (1.f - z) * nn);
      }
      __syncthreads();   // all reads of old h_con complete
#pragma unroll
      for (int i = 0; i < 8; ++i) {
        const int m = i + mo;
        sHc32[m * CON + gcol] = hnew[i];
        sHc16[m * CON + gcol] = (_Float16)hnew[i];
      }
      __syncthreads();
    }

    // phase B: co_params = h_con @ W_co^T + b_co; sample gen input.
    if (wv < 2) {
      v8f acc = {};
      for (int kt = 0; kt < 4; ++kt) {
        v16h a = load_a(sHc16, CON, kt * 32, lid);
        acc = wmma16(a, load_b(pWco, 4, wv, kt, lid), acc);
      }
      const float bv = b_co[wv * 16 + col];
#pragma unroll
      for (int i = 0; i < 8; ++i) {
        const int m = i + mo;
        const float v = acc[i] + bv;
        if (wv == 0) sCoM[m * 16 + col] = v;
        else         sCoS[m * 16 + col] = expf(0.5f * v);
      }
    }
    __syncthreads();
    {
      const int m = tid >> 4, n = tid & 15;  // exactly 256 elements
      const float gi =
          sCoM[m * 16 + n] +
          sCoS[m * 16 + n] * eps[((size_t)(r0 + m) * T_STEPS + t) * CO + n];
      sGi32[m * 16 + n] = gi;
      sGi16[m * 32 + n] = (_Float16)gi;  // cols 16..31 stay zero (K pad)
    }
    __syncthreads();

    // phase C: generator GRU. Wave w owns cols [32w,32w+32) (2 tiles).
    {
      float hs[2][8];
      for (int ct = 0; ct < 2; ++ct) {
        const int nt = 2 * wv + ct;
        v8f az = {}, ar = {}, xn = {}, hn = {};
        {
          v16h a = load_a(sGi16, 32, 0, lid);  // K=16 padded to 32
          az = wmma16(a, load_b(pWihGen, 1, nt, 0, lid), az);
          ar = wmma16(a, load_b(pWihGen, 1, 16 + nt, 0, lid), ar);
          xn = wmma16(a, load_b(pWihGen, 1, 32 + nt, 0, lid), xn);
        }
        for (int kt = 0; kt < 8; ++kt) {
          v16h a = load_a(sHg16, GEN, kt * 32, lid);
          az = wmma16(a, load_b(pWhhGen, 8, nt, kt, lid), az);
          ar = wmma16(a, load_b(pWhhGen, 8, 16 + nt, kt, lid), ar);
          hn = wmma16(a, load_b(pWhhGen, 8, 32 + nt, kt, lid), hn);
        }
        const int gcol = nt * 16 + col;
        const float bz = b_ih_gen[gcol];
        const float br = b_ih_gen[GEN + gcol];
        const float bn = b_ih_gen[2 * GEN + gcol];
#pragma unroll
        for (int i = 0; i < 8; ++i) {
          const int m = i + mo;
          const float hold = sHg32[m * GEN + gcol];
          const float z  = sigm(az[i] + bz);
          const float rr = sigm(ar[i] + br);
          const float nn = tanhf(xn[i] + bn + rr * hn[i]);
          hs[ct][i] = clip5(z * hold + (1.f - z) * nn);
        }
      }
      __syncthreads();   // all reads of old h_gen complete
      for (int ct = 0; ct < 2; ++ct) {
        const int gcol = (2 * wv + ct) * 16 + col;
#pragma unroll
        for (int i = 0; i < 8; ++i) {
          const int m = i + mo;
          sHg32[m * GEN + gcol] = hs[ct][i];
          sHg16[m * GEN + gcol] = (_Float16)hs[ct][i];
        }
      }
      __syncthreads();
    }

    // phase D: factor = h_gen @ W_fac_n^T (also feeds next step's con input)
    if (wv < 4) {
      v8f acc = {};
      for (int kt = 0; kt < 8; ++kt) {
        v16h a = load_a(sHg16, GEN, kt * 32, lid);
        acc = wmma16(a, load_b(pWfac, 8, wv, kt, lid), acc);
      }
      const int gcol = wv * 16 + col;
#pragma unroll
      for (int i = 0; i < 8; ++i) {
        const int m = i + mo;
        sFac[m * FAC + gcol] = acc[i];
        sX16[m * CONIN + CI2 + gcol] = (_Float16)acc[i];
      }
    }
    __syncthreads();

    // phase E: write concatenated outputs for this timestep
    for (int idx = tid; idx < 16 * OUTW; idx += 256) {
      const int m = idx / OUTW;
      const int k = idx - m * OUTW;
      float v;
      if      (k < 256) v = sHg32[m * GEN + k];
      else if (k < 384) v = sHc32[m * CON + (k - 256)];
      else if (k < 400) v = sCoM[m * 16 + (k - 384)];
      else if (k < 416) v = sCoS[m * 16 + (k - 400)];
      else if (k < 432) v = sGi32[m * 16 + (k - 416)];
      else              v = sFac[m * FAC + (k - 432)];
      out[((size_t)(r0 + m) * T_STEPS + t) * OUTW + k] = v;
    }
    __syncthreads();
  }
}

// ---------------------------------------------------------------------------
extern "C" void kernel_launch(void* const* d_in, const int* in_sizes, int n_in,
                              void* d_out, int out_size, void* d_ws,
                              size_t ws_size, hipStream_t stream) {
  const float* ic       = (const float*)d_in[0];
  const float* ci       = (const float*)d_in[1];
  const float* eps      = (const float*)d_in[2];
  const float* W_ic     = (const float*)d_in[3];
  const float* b_ic     = (const float*)d_in[4];
  const float* con_h0   = (const float*)d_in[5];
  const float* W_ih_gen = (const float*)d_in[6];
  const float* b_ih_gen = (const float*)d_in[7];
  const float* W_hh_gen = (const float*)d_in[8];
  const float* W_ih_con = (const float*)d_in[9];
  const float* b_ih_con = (const float*)d_in[10];
  const float* W_hh_con = (const float*)d_in[11];
  const float* W_fac    = (const float*)d_in[12];
  const float* W_co     = (const float*)d_in[13];
  const float* b_co     = (const float*)d_in[14];

  // workspace layout: 64 reciprocal norms, then packed f16 weights (~745 KB)
  float* rnorm = (float*)d_ws;
  _Float16* base = (_Float16*)((char*)d_ws + 256);
  size_t off = 0;
  _Float16* pWic    = base + off; off += (size_t)16 * 2 * 512;  // W_ic    256x64
  _Float16* pWihCon = base + off; off += (size_t)24 * 6 * 512;  // W_ih_con 384x192
  _Float16* pWhhCon = base + off; off += (size_t)24 * 4 * 512;  // W_hh_con 384x128
  _Float16* pWco    = base + off; off += (size_t)2  * 4 * 512;  // W_co     32x128
  _Float16* pWihGen = base + off; off += (size_t)48 * 1 * 512;  // W_ih_gen 768x16->32
  _Float16* pWhhGen = base + off; off += (size_t)48 * 8 * 512;  // W_hh_gen 768x256
  _Float16* pWfac   = base + off; off += (size_t)4  * 8 * 512;  // W_fac_n  64x256

  fac_norm_kernel<<<64, 256, 0, stream>>>(W_fac, rnorm);

  auto pack = [&](const float* W, _Float16* dst, int outDim, int inDim, int KT,
                  const float* rs) {
    int total = (outDim / 16) * KT * 512;
    pack_wt_kernel<<<(total + 255) / 256, 256, 0, stream>>>(W, dst, outDim,
                                                            inDim, KT, rs,
                                                            total);
  };
  pack(W_ic,     pWic,    256, 64,  2, nullptr);
  pack(W_ih_con, pWihCon, 384, 192, 6, nullptr);
  pack(W_hh_con, pWhhCon, 384, 128, 4, nullptr);
  pack(W_co,     pWco,    32,  128, 4, nullptr);
  pack(W_ih_gen, pWihGen, 768, 16,  1, nullptr);  // K zero-padded to 32
  pack(W_hh_gen, pWhhGen, 768, 256, 8, nullptr);
  pack(W_fac,    pWfac,   64,  256, 8, rnorm);    // row-normalized

  lfads_persistent<<<BATCH / 16, 256, 0, stream>>>(
      ic, ci, eps, b_ic, con_h0, b_ih_gen, b_ih_con, b_co, pWic, pWihCon,
      pWhhCon, pWco, pWihGen, pWhhGen, pWfac, (float*)d_out);
}
